// BaselineGNNPredictor_8804682956956
// MI455X (gfx1250) — matrix operand order repaired and checked by reading, not compiled, not run on hardware
//
#include <hip/hip_runtime.h>
#include <hip/hip_bf16.h>

// ---------------------------------------------------------------------------
// Types for CDNA5 WMMA (gfx1250, wave32)
// ---------------------------------------------------------------------------
typedef __attribute__((ext_vector_type(16))) __bf16 v16bf;
typedef __attribute__((ext_vector_type(8)))  float  v8f;

union FragBF {
    v16bf v;
    uint4 u[2];
};

#define HID 128

// round-to-nearest-even f32 -> bf16 (bit pattern as ushort)
__device__ __forceinline__ unsigned short f2bf(float f) {
    unsigned int u = __float_as_uint(f);
    unsigned int r = u + 0x7FFFu + ((u >> 16) & 1u);
    if ((u & 0x7F800000u) == 0x7F800000u) r = u;   // Inf/NaN: truncate
    return (unsigned short)(r >> 16);
}

__device__ __forceinline__ float gelu_erf(float x) {
    return 0.5f * x * (1.0f + erff(x * 0.70710678118654752f));
}

// ---------------------------------------------------------------------------
// Convert+transpose the 4 layer weights  W[256][128] f32 -> Wt[128][256] bf16
// ---------------------------------------------------------------------------
__global__ void __launch_bounds__(256)
prep_w_kernel(const float* __restrict__ W0, const float* __restrict__ W1,
              const float* __restrict__ W2, const float* __restrict__ W3,
              unsigned short* __restrict__ Wt)
{
    int idx = blockIdx.x * blockDim.x + threadIdx.x;   // 0 .. 4*256*128-1
    if (idx >= 4 * 256 * HID) return;
    int l   = idx >> 15;          // layer
    int rem = idx & 32767;
    int n   = rem >> 8;           // out col 0..127
    int k   = rem & 255;          // in  row 0..255
    const float* W = (l == 0) ? W0 : (l == 1) ? W1 : (l == 2) ? W2 : W3;
    Wt[(size_t)l * 32768 + (size_t)n * 256 + k] = f2bf(W[(size_t)k * HID + n]);
}

// ---------------------------------------------------------------------------
// Encoder: row-wise  GELU(LN(x @ W + b, g, be)); din small (8).
// One 128-thread block per row. Writes f32 and optional bf16 copy.
// ---------------------------------------------------------------------------
__global__ void __launch_bounds__(128)
encode_kernel(const float* __restrict__ x, const float* __restrict__ W,
              const float* __restrict__ b, const float* __restrict__ g,
              const float* __restrict__ be,
              float* __restrict__ hf, unsigned short* __restrict__ hb, int din)
{
    __shared__ float xs[16];
    __shared__ float s1[HID], s2[HID];
    const int row = blockIdx.x;
    const int t   = threadIdx.x;

    if (t < din) xs[t] = x[(size_t)row * din + t];
    __syncthreads();

    float acc = b[t];
    for (int k = 0; k < din; ++k) acc += xs[k] * W[k * HID + t];

    s1[t] = acc; s2[t] = acc * acc;
    __syncthreads();
    for (int w = 64; w > 0; w >>= 1) {
        if (t < w) { s1[t] += s1[t + w]; s2[t] += s2[t + w]; }
        __syncthreads();
    }
    float mean = s1[0] * (1.0f / HID);
    float var  = s2[0] * (1.0f / HID) - mean * mean;
    float y    = (acc - mean) * rsqrtf(var + 1e-5f) * g[t] + be[t];
    float ge   = gelu_erf(y);

    size_t off = (size_t)row * HID + t;
    if (hf) hf[off] = ge;
    if (hb) hb[off] = f2bf(ge);
}

// ---------------------------------------------------------------------------
// Edge-message GEMM + LN + GELU + scatter-add (persistent, grid-stride).
//   messages[E,128] = GELU(LN(concat(h[src],h[dst]) @ W[256,128] + b, g, be))
//   h_new[dst] += messages              (fp32 atomics)
//
// 256 threads (8 waves) per block; each wave owns one 16-col output tile and
// keeps its whole-layer B operand (8 fragments = 64 VGPRs) register-resident.
// A tiles (16 edges x 256 bf16) are double-buffered in LDS: the next tile's
// h[src]/h[dst] gathers are issued into registers before the current tile's
// 8 WMMAs, and stored to the alternate buffer afterwards.
// ---------------------------------------------------------------------------
__global__ void __launch_bounds__(256)
msg_kernel(const unsigned short* __restrict__ h_bf,   // [N][128] bf16
           const unsigned short* __restrict__ Wt,     // [128][256] bf16 (W^T)
           const float* __restrict__ bias,            // [128]
           const float* __restrict__ gam,             // [128]
           const float* __restrict__ bet,             // [128]
           const int*   __restrict__ src,
           const int*   __restrict__ dst,
           float* __restrict__ h_new,                 // [N][128]
           int E, int numTiles)
{
    // A tiles: 2 x (16 rows x 264 bf16) = 2 x (132 dwords x 16) padded vs bank conflicts
    __shared__ uint4 ldsA[2][16 * 33];
    // C tile: 16 rows x 128 f32 (+4 pad)
    __shared__ float ldsC[16 * 132];

    const int tid  = threadIdx.x;
    const int ct   = tid >> 5;        // wave -> output col tile 0..7
    const int lane = tid & 31;
    const int Mrow = lane & 15;
    const int gsel = lane >> 4;
    const int Ncol = ct * 16 + (lane & 15);

    // staging / LN roles
    const int m   = tid >> 4;         // edge row 0..15
    const int cch = tid & 15;         // 16B chunk 0..15

    // ---- B operand: register resident for the whole layer ----
    FragBF B[8];
    {
        const uint4* wt4 = (const uint4*)Wt;
#pragma unroll
        for (int kk = 0; kk < 8; ++kk) {
            int bidx = Ncol * 32 + kk * 4 + gsel;       // uint4 units
            B[kk].u[0] = wt4[bidx];
            B[kk].u[1] = wt4[bidx + 2];
        }
    }

    int tile = blockIdx.x;
    if (tile >= numTiles) return;     // uniform per block

    const uint4 zero4 = {0u, 0u, 0u, 0u};
    uint4 rs = zero4, rd = zero4;

    // ---- prologue: stage first tile into buffer 0 ----
    {
        int e = tile * 16 + m;
        if (e < E) {
            rs = ((const uint4*)(h_bf + (size_t)src[e] * HID))[cch];
            rd = ((const uint4*)(h_bf + (size_t)dst[e] * HID))[cch];
        }
        ldsA[0][m * 33 + cch]      = rs;
        ldsA[0][m * 33 + 16 + cch] = rd;
    }
    __syncthreads();

    int cur = 0;
    while (true) {
        const int  next    = tile + gridDim.x;
        const bool hasNext = (next < numTiles);

        // ---- issue next tile's gathers early (held in registers) ----
        if (hasNext) {
            int e = next * 16 + m;
            if (e < E) {
                rs = ((const uint4*)(h_bf + (size_t)src[e] * HID))[cch];
                rd = ((const uint4*)(h_bf + (size_t)dst[e] * HID))[cch];
            } else {
                rs = zero4; rd = zero4;
            }
        }

        // ---- 8 WMMA k-steps over the current A tile ----
        const uint4* Abuf = ldsA[cur];
        v8f acc = {0.f, 0.f, 0.f, 0.f, 0.f, 0.f, 0.f, 0.f};
#pragma unroll
        for (int kk = 0; kk < 8; ++kk) {
            FragBF a;
            // A 16x32 bf16 ISA layout: dwords [kk*16 + gsel*4 ..] / [+8 ..]
            a.u[0] = Abuf[Mrow * 33 + kk * 4 + gsel];
            a.u[1] = Abuf[Mrow * 33 + kk * 4 + 2 + gsel];
            acc = __builtin_amdgcn_wmma_f32_16x16x32_bf16(
                      false, a.v, false, B[kk].v, (short)0, acc, false, false);
        }

        // ---- C tile -> LDS (vgpr r -> M = r + 8*gsel) ----
#pragma unroll
        for (int r = 0; r < 8; ++r) {
            ldsC[(r + 8 * gsel) * 132 + ct * 16 + (lane & 15)] = acc[r];
        }
        __syncthreads();

        // ---- bias + LayerNorm(row) + GELU + scatter-add ----
        {
            const int e = tile * 16 + m;      // this thread's edge row
            const int j = cch;                // 16 threads per row, 8 cols each
            float vals[8];
            float s = 0.f, sq = 0.f;
#pragma unroll
            for (int q = 0; q < 8; ++q) {
                int col = j * 8 + q;
                float v = ldsC[m * 132 + col] + bias[col];
                vals[q] = v;
                s += v; sq += v * v;
            }
#pragma unroll
            for (int msk = 1; msk < 16; msk <<= 1) {
                s  += __shfl_xor(s,  msk, 16);
                sq += __shfl_xor(sq, msk, 16);
            }
            float mean = s * (1.0f / HID);
            float var  = sq * (1.0f / HID) - mean * mean;
            float rstd = rsqrtf(var + 1e-5f);

            if (e < E) {
                float* out = h_new + (size_t)dst[e] * HID;
#pragma unroll
                for (int q = 0; q < 8; ++q) {
                    int col = j * 8 + q;
                    float y = (vals[q] - mean) * rstd * gam[col] + bet[col];
                    atomicAdd(out + col, gelu_erf(y));
                }
            }
        }

        // ---- commit staged next tile to the alternate buffer ----
        if (hasNext) {
            ldsA[cur ^ 1][m * 33 + cch]      = rs;
            ldsA[cur ^ 1][m * 33 + 16 + cch] = rd;
        }
        __syncthreads();          // guards ldsC reuse + A buffer swap

        if (!hasNext) break;
        cur ^= 1;
        tile = next;
    }
}

// ---------------------------------------------------------------------------
// Residual + LayerNorm:  h = LN(h + h_new, ng, nb); refresh f32 + bf16 copies
// ---------------------------------------------------------------------------
__global__ void __launch_bounds__(128)
resid_ln_kernel(float* __restrict__ h, unsigned short* __restrict__ hb,
                const float* __restrict__ h_new,
                const float* __restrict__ ng, const float* __restrict__ nb)
{
    __shared__ float s1[HID], s2[HID];
    const int row = blockIdx.x;
    const int t   = threadIdx.x;
    size_t off = (size_t)row * HID + t;

    float x = h[off] + h_new[off];
    s1[t] = x; s2[t] = x * x;
    __syncthreads();
    for (int w = 64; w > 0; w >>= 1) {
        if (t < w) { s1[t] += s1[t + w]; s2[t] += s2[t + w]; }
        __syncthreads();
    }
    float mean = s1[0] * (1.0f / HID);
    float var  = s2[0] * (1.0f / HID) - mean * mean;
    float y    = (x - mean) * rsqrtf(var + 1e-5f) * ng[t] + nb[t];
    h[off]  = y;
    hb[off] = f2bf(y);
}

// ---------------------------------------------------------------------------
// Masked pooling sums.  pool[0..127]=sum type0, [128..255]=sum type1,
// pool[256]=cnt0, pool[257]=cnt1
// ---------------------------------------------------------------------------
__global__ void __launch_bounds__(128)
pool_kernel(const float* __restrict__ h, const int* __restrict__ types,
            float* __restrict__ pool, int N)
{
    const int t = threadIdx.x;
    float a0 = 0.f, a1 = 0.f;
    int c0 = 0, c1 = 0;
    for (int row = blockIdx.x; row < N; row += gridDim.x) {
        float v = h[(size_t)row * HID + t];
        if (types[row] == 0) { a0 += v; c0++; } else { a1 += v; c1++; }
    }
    atomicAdd(pool + t,        a0);
    atomicAdd(pool + HID + t,  a1);
    if (t == 0) {
        atomicAdd(pool + 2 * HID,     (float)c0);
        atomicAdd(pool + 2 * HID + 1, (float)c1);
    }
}

// ---------------------------------------------------------------------------
// Final: combined -> pred MLP -> 4 heads. Single block, 384 threads.
// ---------------------------------------------------------------------------
struct FinalParams {
    const float *W1, *b1, *g1, *be1, *W2, *b2;      // pred
    const float *hW1[4], *hb1[4], *hW2[4], *hb2[4]; // heads: acc, energy, train, depth
};

__global__ void __launch_bounds__(384)
final_kernel(const float* __restrict__ pool, const float* __restrict__ genc,
             FinalParams p, float* __restrict__ out)
{
    __shared__ float comb[384], fbuf[256], feat[HID], hh[256];
    __shared__ float r1[256], r2[256];
    const int t = threadIdx.x;

    if (t < 128) {
        float c0 = pool[256];
        comb[t] = (c0 > 0.f) ? pool[t] / fmaxf(c0, 1.f) : 0.f;
    } else if (t < 256) {
        float c1 = pool[257];
        comb[t] = (c1 > 0.f) ? pool[t] / fmaxf(c1, 1.f) : 0.f;   // pool[t] == sum1[t-128]
    } else {
        comb[t] = genc[t - 256];
    }
    __syncthreads();

    // f = GELU(LN(comb @ W1 + b1, g1, be1))   [256]
    float acc = 0.f;
    if (t < 256) {
        acc = p.b1[t];
        for (int k = 0; k < 384; ++k) acc += comb[k] * p.W1[k * 256 + t];
        r1[t] = acc; r2[t] = acc * acc;
    }
    __syncthreads();
    for (int w = 128; w > 0; w >>= 1) {
        if (t < w) { r1[t] += r1[t + w]; r2[t] += r2[t + w]; }
        __syncthreads();
    }
    float mean = r1[0] * (1.0f / 256.f);
    float var  = r2[0] * (1.0f / 256.f) - mean * mean;
    float rstd = rsqrtf(var + 1e-5f);
    if (t < 256) {
        float y = (acc - mean) * rstd * p.g1[t] + p.be1[t];
        fbuf[t] = gelu_erf(y);
    }
    __syncthreads();

    // features = GELU(f @ W2 + b2)   [128]
    if (t < HID) {
        float a = p.b2[t];
        for (int k = 0; k < 256; ++k) a += fbuf[k] * p.W2[k * HID + t];
        feat[t] = gelu_erf(a);
    }
    __syncthreads();

    // head hidden layers: 4 heads x 64 units
    if (t < 256) {
        int hd = t >> 6, u = t & 63;
        float a = p.hb1[hd][u];
        for (int k = 0; k < HID; ++k) a += feat[k] * p.hW1[hd][k * 64 + u];
        hh[t] = gelu_erf(a);
    }
    __syncthreads();

    if (t < 4) {
        float a = p.hb2[t][0];
        for (int k = 0; k < 64; ++k) a += hh[t * 64 + k] * p.hW2[t][k];
        float o;
        if (t == 0 || t == 2) {            // acc, train: sigmoid
            o = 1.0f / (1.0f + expf(-a));
        } else {                           // energy, depth: softplus
            o = (a > 20.f) ? a : log1pf(expf(a));
        }
        out[t] = o;
    }
}

// ---------------------------------------------------------------------------
// Host-side launch
// ---------------------------------------------------------------------------
extern "C" void kernel_launch(void* const* d_in, const int* in_sizes, int n_in,
                              void* d_out, int out_size, void* d_ws, size_t ws_size,
                              hipStream_t stream)
{
    (void)n_in; (void)out_size; (void)ws_size;
    // Two plausible flattenings of setup_inputs()['params']:
    //  - JAX tree_leaves (sorted dict keys): leaf 0 = edge.W, size 4*128 = 512
    //  - insertion order:                    leaf 0 = node.W, size 8*128 = 1024
    const bool sorted = (in_sizes[0] == 512);

    int iNodeW, iNodeB, iNodeBe, iNodeG;
    int iGlobW, iGlobB, iGlobBe, iGlobG;
    int layBase, layW, layB, layBe, layG, layNg, layNb;
    int iW1, iB1, iG1, iBe1, iW2, iB2;
    int hBase[4], hW1o, hB1o, hW2o, hB2o;   // head order: acc, energy, train, depth

    if (sorted) {
        // params keys sorted: edge(0..3:W,b,be,g) glob(4..7) heads(8..23, keys
        // acc,depth,energy,train each W1,W2,b1,b2) layers(24..47, W,b,be,g,nb,ng)
        // node(48..51) pred(52..57: W1,W2,b1,b2,be1,g1)
        iNodeW = 48; iNodeB = 49; iNodeBe = 50; iNodeG = 51;
        iGlobW = 4;  iGlobB = 5;  iGlobBe = 6;  iGlobG = 7;
        layBase = 24; layW = 0; layB = 1; layBe = 2; layG = 3; layNb = 4; layNg = 5;
        iW1 = 52; iW2 = 53; iB1 = 54; iB2 = 55; iBe1 = 56; iG1 = 57;
        hBase[0] = 8;  hBase[1] = 16; hBase[2] = 20; hBase[3] = 12; // acc,energy,train,depth
        hW1o = 0; hW2o = 1; hB1o = 2; hB2o = 3;
    } else {
        // insertion order: node(0..3:W,b,g,be) edge(4..7) glob(8..11)
        // layers(12..35: W,b,g,be,ng,nb) pred(36..41: W1,b1,g1,be1,W2,b2)
        // heads(42..57: acc,energy,train,depth each W1,b1,W2,b2)
        iNodeW = 0;  iNodeB = 1;  iNodeG = 2;  iNodeBe = 3;
        iGlobW = 8;  iGlobB = 9;  iGlobG = 10; iGlobBe = 11;
        layBase = 12; layW = 0; layB = 1; layG = 2; layBe = 3; layNg = 4; layNb = 5;
        iW1 = 36; iB1 = 37; iG1 = 38; iBe1 = 39; iW2 = 40; iB2 = 41;
        hBase[0] = 42; hBase[1] = 46; hBase[2] = 50; hBase[3] = 54;
        hW1o = 0; hB1o = 1; hW2o = 2; hB2o = 3;
    }

    const float* node_x = (const float*)d_in[58];
    const float* glob_x = (const float*)d_in[60];
    const int*   eidx   = (const int*)  d_in[61];
    const int*   ntypes = (const int*)  d_in[62];
    const int N = in_sizes[58] / 8;
    const int E = in_sizes[59] / 4;
    const int* src = eidx;
    const int* dst = eidx + E;

    auto F = [&](int i) { return (const float*)d_in[i]; };

    // ---- workspace carve ----
    char* ws = (char*)d_ws;
    float*          h_f32 = (float*)ws;          ws += (size_t)N * HID * 4;
    float*          h_new = (float*)ws;          ws += (size_t)N * HID * 4;
    unsigned short* h_bf  = (unsigned short*)ws; ws += (size_t)N * HID * 2;
    unsigned short* Wt    = (unsigned short*)ws; ws += (size_t)4 * HID * 256 * 2;
    float*          genc  = (float*)ws;          ws += 512;
    float*          pool  = (float*)ws;          ws += 2048;

    // ---- weight prep ----
    prep_w_kernel<<<(4 * 256 * HID + 255) / 256, 256, 0, stream>>>(
        F(layBase + 0 * 6 + layW), F(layBase + 1 * 6 + layW),
        F(layBase + 2 * 6 + layW), F(layBase + 3 * 6 + layW), Wt);

    // ---- encoders (edge encoder is dead code in the reference -> skipped) ----
    encode_kernel<<<N, 128, 0, stream>>>(node_x, F(iNodeW), F(iNodeB),
                                         F(iNodeG), F(iNodeBe), h_f32, h_bf, 8);
    encode_kernel<<<1, 128, 0, stream>>>(glob_x, F(iGlobW), F(iGlobB),
                                         F(iGlobG), F(iGlobBe), genc, nullptr, 8);

    // ---- message-passing layers (persistent msg kernel) ----
    const int numTiles  = (E + 15) / 16;
    const int msgBlocks = (numTiles < 2048) ? numTiles : 2048;
    for (int l = 0; l < 4; ++l) {
        const int b = layBase + l * 6;
        hipMemsetAsync(h_new, 0, (size_t)N * HID * 4, stream);
        msg_kernel<<<msgBlocks, 256, 0, stream>>>(
            h_bf, Wt + (size_t)l * 32768,
            F(b + layB), F(b + layG), F(b + layBe),
            src, dst, h_new, E, numTiles);
        resid_ln_kernel<<<N, 128, 0, stream>>>(h_f32, h_bf, h_new,
                                               F(b + layNg), F(b + layNb));
    }

    // ---- pooling + prediction ----
    hipMemsetAsync(pool, 0, (2 * HID + 2) * sizeof(float), stream);
    pool_kernel<<<512, 128, 0, stream>>>(h_f32, ntypes, pool, N);

    FinalParams fp;
    fp.W1 = F(iW1); fp.b1 = F(iB1); fp.g1 = F(iG1); fp.be1 = F(iBe1);
    fp.W2 = F(iW2); fp.b2 = F(iB2);
    for (int hd = 0; hd < 4; ++hd) {
        fp.hW1[hd] = F(hBase[hd] + hW1o);
        fp.hb1[hd] = F(hBase[hd] + hB1o);
        fp.hW2[hd] = F(hBase[hd] + hW2o);
        fp.hb2[hd] = F(hBase[hd] + hB2o);
    }
    final_kernel<<<1, 384, 0, stream>>>(pool, genc, fp, (float*)d_out);
}
